// Somatt_25512105738273
// MI455X (gfx1250) — matrix operand order repaired and checked by reading, not compiled, not run on hardware
//
#include <hip/hip_runtime.h>
#include <hip/hip_bf16.h>
#include <math.h>

// ---------------------------------------------------------------------------
// Somatic-mutation transformer for MI455X (gfx1250), wave32 + WMMA f16.
//
// All GEMMs use v_wmma_f32_16x16x32_f16 with 32x64 register blocking per wave
// (2 A-frags x 4 B-frags -> 8 accumulators, 8 WMMAs per K-step).  B fragments
// are loaded into distinct registers up front each K-step so the four loads
// issue as one clause and the eight WMMAs stream back-to-back.  Weights are
// converted f32->f16 and transposed to [N,K] once per launch so each lane's
// B-fragment is one contiguous 32B load (B order used here: lane (n,g) holds
// K=16g+h, h=0..15).  A fragments use the same order: lane (m,g):
// K=16g+h, h=0..15.  C/D layout: lane n=lane%16,
// rows r+8*(lane/16).
//
// d_in layout assumption: inputs in setup_inputs() dict order (17 arrays),
// then `params` flattened as a JAX pytree (dict keys sorted, lists/tuples in
// order) -> 80 leaves, n_in = 97.  Indices below follow that order.
// ---------------------------------------------------------------------------

typedef _Float16 v16h __attribute__((ext_vector_type(16)));
typedef float    v8f  __attribute__((ext_vector_type(8)));

#define Bc    16
#define Dd    256
#define Hh    8
#define NLn   4
#define Sq    1286      // 6 + 768 + 512
#define SpQ   1312      // Sq padded to multiple of 32 for attention tiles
#define NIg   768       // gene pool slots (512 gene + 256 focal)
#define NSs   512       // seg pool slots
#define GPAD  20000
#define SPAD  5000
#define MR    (Bc*Sq)   // 20576 rows, multiple of 32

static __device__ __forceinline__ v8f wmma_f16(v16h a, v16h b, v8f c) {
  return __builtin_amdgcn_wmma_f32_16x16x32_f16(false, a, false, b, (short)0, c,
                                                false, false);
}

// --------------------------- weight convert/transpose ----------------------
// in: f32 [K,N] row-major -> out: f16 [N,K] row-major
__global__ void cvtT_kernel(const float* __restrict__ W, _Float16* __restrict__ Wt,
                            int K, int N) {
  int idx = blockIdx.x * 256 + threadIdx.x;
  if (idx >= K * N) return;
  int k = idx / N, n = idx % N;
  Wt[(size_t)n * K + k] = (_Float16)W[idx];
}

// --------------------------- 32x64 WMMA GEMM --------------------------------
// C[M,N] = act(A[M,K] @ W[K,N] + bias) + res     (act: 0=none, 1=tanh-gelu)
// Wt is W transposed+f16: [N,K].  One wave per 32x64 macro-tile.
// Requires M%32==0, N%64==0, K%32==0.
__global__ void gemm16_kernel(const float* __restrict__ A,
                              const _Float16* __restrict__ Wt,
                              const float* __restrict__ bias,
                              const float* __restrict__ res,
                              float* __restrict__ C,
                              int M, int N, int K, int act) {
  int lane = threadIdx.x & 31;
  int g = lane >> 4, ln = lane & 15;
  int n0 = blockIdx.x * 64, m0 = blockIdx.y * 32;

  v8f acc[8];
#pragma unroll
  for (int i = 0; i < 8; i++) acc[i] = (v8f){};

  const float* arow0 = A + (size_t)(m0 + ln) * K;        // rows m0..m0+15
  const float* arow1 = A + (size_t)(m0 + 16 + ln) * K;   // rows m0+16..m0+31
  const _Float16* bp[4];
#pragma unroll
  for (int tn = 0; tn < 4; tn++)
    bp[tn] = Wt + (size_t)(n0 + 16 * tn + ln) * K + 16 * g;

  for (int k0 = 0; k0 < K; k0 += 32) {
    // prefetch next A K-step into cache (global_prefetch_b8)
    __builtin_prefetch(arow0 + k0 + 32 + 8 * g, 0, 0);
    __builtin_prefetch(arow1 + k0 + 32 + 8 * g, 0, 0);

    // load all four B fragments first (distinct regs -> one load clause)
    v16h bf[4];
#pragma unroll
    for (int tn = 0; tn < 4; tn++)
      bf[tn] = *(const v16h*)(bp[tn] + k0);  // 32B aligned: K%32==0, 16g halves

    union { v16h v; _Float16 e[16]; } a0, a1;
    const float* p00 = arow0 + k0 + 16 * g;  // K = 16g..16g+7
    const float* p01 = p00 + 8;              // K = 16g+8..16g+15
    const float* p10 = arow1 + k0 + 16 * g;
    const float* p11 = p10 + 8;
#pragma unroll
    for (int t = 0; t < 8; t++) {
      a0.e[t] = (_Float16)p00[t]; a0.e[8 + t] = (_Float16)p01[t];
      a1.e[t] = (_Float16)p10[t]; a1.e[8 + t] = (_Float16)p11[t];
    }
#pragma unroll
    for (int tn = 0; tn < 4; tn++) {
      acc[tn]     = wmma_f16(a0.v, bf[tn], acc[tn]);
      acc[4 + tn] = wmma_f16(a1.v, bf[tn], acc[4 + tn]);
    }
  }

#pragma unroll
  for (int tm = 0; tm < 2; tm++) {
#pragma unroll
    for (int tn = 0; tn < 4; tn++) {
      v8f a = acc[4 * tm + tn];
      int col = n0 + 16 * tn + ln;
      float bv = bias ? bias[col] : 0.f;
#pragma unroll
      for (int r = 0; r < 8; r++) {
        int row = m0 + 16 * tm + r + 8 * g;
        float v = a[r] + bv;
        if (act == 1) {
          float x = v, c = x + 0.044715f * x * x * x;
          v = 0.5f * x * (1.f + tanhf(0.7978845608028654f * c));
        }
        if (res) v += res[(size_t)row * N + col];
        C[(size_t)row * N + col] = v;
      }
    }
  }
}

// --------------------------- LayerNorm -------------------------------------
__global__ void ln_kernel(const float* __restrict__ in, const float* __restrict__ sc,
                          const float* __restrict__ bi, float* __restrict__ out,
                          const int* __restrict__ pad) {
  __shared__ float red[256];
  int row = blockIdx.x, d = threadIdx.x;
  float v = in[(size_t)row * 256 + d];
  red[d] = v; __syncthreads();
  for (int s = 128; s > 0; s >>= 1) { if (d < s) red[d] += red[d + s]; __syncthreads(); }
  float mu = red[0] * (1.f / 256.f); __syncthreads();
  float c = v - mu; red[d] = c * c; __syncthreads();
  for (int s = 128; s > 0; s >>= 1) { if (d < s) red[d] += red[d + s]; __syncthreads(); }
  float var = red[0] * (1.f / 256.f);
  float y = c * rsqrtf(var + 1e-5f) * sc[d] + bi[d];
  if (pad && pad[row]) y = 0.f;
  out[(size_t)row * 256 + d] = y;
}

// --------------------------- feature builders -------------------------------
// F [B*768, 320] : [:256]=gene_emb|fc_gene_emb, [256:]=maf-proj|focal-cna-emb
__global__ void features_kernel(const float* __restrict__ gene_emb,
                                const float* __restrict__ maf,
                                const float* __restrict__ fc_emb,
                                const int*   __restrict__ fc_alt,
                                const float* __restrict__ W_maf,
                                const float* __restrict__ b_maf,
                                const float* __restrict__ emb_focal,
                                float* __restrict__ F) {
  int j = blockIdx.x, b = blockIdx.y, t = threadIdx.x;
  float v;
  if (t < 256) {
    v = (j < 512) ? gene_emb[((size_t)(b * 512 + j)) * 256 + t]
                  : fc_emb[((size_t)(b * 256 + (j - 512))) * 256 + t];
  } else {
    int dm = t - 256;
    if (j < 512) {
      float m = maf[b * 512 + j];
      if (__builtin_isnan(m)) m = 0.f;
      v = m * W_maf[dm] + b_maf[dm];
    } else {
      v = emb_focal[fc_alt[b * 256 + (j - 512)] * 64 + dm];
    }
  }
  F[((size_t)(b * NIg + j)) * 320 + t] = v;
}

// seg_emb [B*512,256] = [start,end,mean] @ W_seg[3,256] + b_seg
__global__ void segproj_kernel(const float* __restrict__ ss, const float* __restrict__ se,
                               const float* __restrict__ sm, const float* __restrict__ W,
                               const float* __restrict__ bs, float* __restrict__ out) {
  int j = blockIdx.x, b = blockIdx.y, d = threadIdx.x;
  float a = ss[b * 512 + j], e = se[b * 512 + j], m = sm[b * 512 + j];
  out[((size_t)(b * NSs + j)) * 256 + d] =
      a * W[d] + e * W[256 + d] + m * W[512 + d] + bs[d];
}

// --------------------------- mean-pool by unique id -------------------------
// One block (256 thr) per batch row: bitonic sort (id,idx) in LDS, segment
// index, counts, then serial segment-mean with 256-wide vector lanes.
__global__ void pool_kernel(const float* __restrict__ e,
                            const int* __restrict__ ids0, const int* __restrict__ ids1,
                            int n0, int n1, int npow, int padval,
                            float* __restrict__ out, int* __restrict__ uids) {
  __shared__ int sid[1024], sidx[1024], sseg[1024], scnt[1024];
  int b = blockIdx.x, tid = threadIdx.x;
  int N = n0 + n1;

  for (int i = tid; i < npow; i += 256) {
    int v = 0x7fffffff;
    if (i < n0)      v = ids0[b * n0 + i];
    else if (i < N)  v = ids1[b * n1 + (i - n0)];
    sid[i] = v; sidx[i] = i;
  }
  // bitonic sort (non-stable; fine: equal ids pool identically)
  for (int k = 2; k <= npow; k <<= 1)
    for (int j = k >> 1; j > 0; j >>= 1) {
      __syncthreads();
      for (int i = tid; i < npow; i += 256) {
        int ixj = i ^ j;
        if (ixj > i) {
          int a = sid[i], bb = sid[ixj];
          bool up = ((i & k) == 0);
          if (up ? (a > bb) : (a < bb)) {
            sid[i] = bb; sid[ixj] = a;
            int t = sidx[i]; sidx[i] = sidx[ixj]; sidx[ixj] = t;
          }
        }
      }
    }
  __syncthreads();
  if (tid == 0) {                     // seg = cumsum(is_new) - 1
    int s = -1;
    for (int j = 0; j < N; j++) {
      if (j == 0 || sid[j] != sid[j - 1]) s++;
      sseg[j] = s;
    }
  }
  __syncthreads();
  for (int i = tid; i < npow; i += 256) scnt[i] = 0;
  for (int i = tid; i < N; i += 256) uids[b * N + i] = padval;
  __syncthreads();
  for (int j = tid; j < N; j += 256) {
    atomicAdd(&scnt[sseg[j]], 1);
    uids[b * N + sseg[j]] = sid[j];   // same value per segment; race-safe
  }
  __syncthreads();
  // segment means: thread tid owns channel d=tid, walk sorted rows serially
  int d = tid;
  float acc = 0.f; int cur = 0;
  for (int j = 0; j < N; j++) {
    if (sseg[j] != cur) {
      out[((size_t)(b * N + cur)) * 256 + d] = acc / (float)scnt[cur];
      acc = 0.f; cur = sseg[j];
    }
    acc += e[((size_t)(b * N + sidx[j])) * 256 + d];
  }
  out[((size_t)(b * N + cur)) * 256 + d] = acc / (float)scnt[cur];
}

// --------------------------- token assembly --------------------------------
__global__ void assemble_kernel(const int* cancer, const int* cand, const int* sex,
                                const float* age, const int* race, const float* timev,
                                const float* eventv,
                                const float* emb_cancer, const float* emb_cancer_d,
                                const float* emb_sex, const float* emb_race,
                                const float* W_age, const float* b_age,
                                const float* W_surv, const float* b_surv,
                                const float* emb_clin_id, const float* emb_gene_id,
                                const float* emb_seg_id,
                                const float* poolG, const int* uidG,
                                const float* poolS, const int* uidS,
                                float* __restrict__ x, int* __restrict__ pad) {
  int s = blockIdx.x, b = blockIdx.y, d = threadIdx.x;
  float val = 0.f, pos = 0.f; int pd = 0;
  if (s < 6) {
    switch (s) {
      case 0: val = emb_cancer[cancer[b] * 256 + d]; break;
      case 1: val = emb_cancer_d[cand[b] * 256 + d]; break;
      case 2: { int sx = sex[b]; val = emb_sex[sx * 256 + d]; pd = (sx == 2); } break;
      case 3: { float a = age[b]; pd = __builtin_isnan(a); if (pd) a = 0.f;
                val = a * W_age[d] + b_age[d]; } break;
      case 4: { int rc = race[b]; val = emb_race[rc * 256 + d]; pd = (rc == 5); } break;
      case 5: { float tt = timev[b]; pd = __builtin_isnan(tt); if (pd) tt = 0.f;
                val = tt * W_surv[d] + eventv[b] * W_surv[256 + d] + b_surv[d]; } break;
    }
    pos = emb_clin_id[s * 256 + d];
  } else if (s < 6 + NIg) {
    int gi = s - 6; int uid = uidG[b * NIg + gi]; pd = (uid == GPAD);
    val = poolG[((size_t)(b * NIg + gi)) * 256 + d];
    pos = emb_gene_id[(size_t)uid * 256 + d];
  } else {
    int si = s - 6 - NIg; int uid = uidS[b * NSs + si]; pd = (uid == SPAD);
    val = poolS[((size_t)(b * NSs + si)) * 256 + d];
    pos = emb_seg_id[(size_t)uid * 256 + d];
  }
  if (pd) val = 0.f;
  x[((size_t)(b * Sq + s)) * 256 + d] = val + pos;
  if (d == 0) pad[b * Sq + s] = pd;
}

// --------------------------- QKV repack to f16 ------------------------------
// qkv [B*Sq,768] f32 -> Qh,Kh [B,H,SpQ,32] f16 (zero-padded rows),
//                       Vt    [B,H,32,SpQ] f16 (transposed for B-frag loads)
__global__ void repack_kernel(const float* __restrict__ qkv,
                              _Float16* __restrict__ Qh, _Float16* __restrict__ Kh,
                              _Float16* __restrict__ Vt) {
  int s = blockIdx.x, b = blockIdx.y, d = threadIdx.x;  // d in [0,32)
  bool valid = s < Sq;
  for (int h = 0; h < Hh; h++) {
    float q = 0.f, k = 0.f, v = 0.f;
    if (valid) {
      const float* row = qkv + ((size_t)(b * Sq + s)) * 768 + h * 32 + d;
      q = row[0]; k = row[256]; v = row[512];
    }
    size_t base = ((size_t)(b * Hh + h) * SpQ + s) * 32 + d;
    Qh[base] = (_Float16)q;
    Kh[base] = (_Float16)k;
    Vt[((size_t)(b * Hh + h) * 32 + d) * SpQ + s] = (_Float16)v;
  }
}

// --------------------------- flash attention --------------------------------
// One wave per (q-tile of 16, head, batch).  dh=32 == WMMA K exactly.
__global__ void attn_kernel(const _Float16* __restrict__ Qh,
                            const _Float16* __restrict__ Kh,
                            const _Float16* __restrict__ Vt,
                            const int* __restrict__ pad,
                            float* __restrict__ out) {
  __shared__ _Float16 pbuf[16 * 32];   // P tile staging: C-layout -> A-layout
  int qt = blockIdx.x, hh = blockIdx.y, b = blockIdx.z;
  int lane = threadIdx.x, g = lane >> 4, ln = lane & 15;

  const _Float16* Qb = Qh + ((size_t)(b * Hh + hh) * SpQ) * 32;
  const _Float16* Kb = Kh + ((size_t)(b * Hh + hh) * SpQ) * 32;
  const _Float16* Vb = Vt + ((size_t)(b * Hh + hh) * 32) * SpQ;

  union { v16h v; _Float16 e[16]; } aq;
  {
    const _Float16* qrow = Qb + (size_t)(qt * 16 + ln) * 32;
#pragma unroll
    for (int t = 0; t < 8; t++) { aq.e[t] = qrow[16 * g + t]; aq.e[8 + t] = qrow[16 * g + 8 + t]; }
  }

  v8f O0 = {}, O1 = {};
  float mrow[8], lrow[8];
#pragma unroll
  for (int r = 0; r < 8; r++) { mrow[r] = -1e30f; lrow[r] = 0.f; }
  const float scale = 0.17677669529663687f;  // 1/sqrt(32)
  v8f zero = {};

  for (int k0 = 0; k0 < Sq; k0 += 32) {
    __builtin_prefetch(Kb + (size_t)(k0 + 32 + ln) * 32, 0, 0);
    // distinct regs: K tiles + V tiles can all load in one clause
    v16h kb0 = *(const v16h*)(Kb + (size_t)(k0 + ln) * 32 + 16 * g);
    v16h kb1 = *(const v16h*)(Kb + (size_t)(k0 + 16 + ln) * 32 + 16 * g);
    v16h vb0 = *(const v16h*)(Vb + (size_t)(ln) * SpQ + k0 + 16 * g);       // d cols 0-15
    v16h vb1 = *(const v16h*)(Vb + (size_t)(16 + ln) * SpQ + k0 + 16 * g);  // d cols 16-31
    v8f s0 = wmma_f16(aq.v, kb0, zero);
    v8f s1 = wmma_f16(aq.v, kb1, zero);

    int c0 = k0 + ln, c1 = k0 + 16 + ln;
    bool m0 = (c0 >= Sq) || pad[b * Sq + c0];
    bool m1 = (c1 >= Sq) || pad[b * Sq + c1];

    float tmax[8];
#pragma unroll
    for (int r = 0; r < 8; r++) {
      float v0 = m0 ? -1e9f : s0[r] * scale;
      float v1 = m1 ? -1e9f : s1[r] * scale;
      s0[r] = v0; s1[r] = v1;
      tmax[r] = fmaxf(v0, v1);
    }
#pragma unroll
    for (int r = 0; r < 8; r++) {        // row max across the 16-lane group
      float v = tmax[r];
      for (int off = 8; off; off >>= 1) v = fmaxf(v, __shfl_xor(v, off, 16));
      tmax[r] = v;
    }
#pragma unroll
    for (int r = 0; r < 8; r++) {        // online softmax update
      float mn = fmaxf(mrow[r], tmax[r]);
      float alpha = __expf(mrow[r] - mn);
      mrow[r] = mn;
      s0[r] = __expf(s0[r] - mn);
      s1[r] = __expf(s1[r] - mn);
      O0[r] *= alpha; O1[r] *= alpha;
      float ls = s0[r] + s1[r];
      for (int off = 8; off; off >>= 1) ls += __shfl_xor(ls, off, 16);
      lrow[r] = lrow[r] * alpha + ls;
    }
    __syncthreads();                      // pbuf free from previous iter
#pragma unroll
    for (int r = 0; r < 8; r++) {         // C-layout write: row r+8g, col ln / 16+ln
      pbuf[(r + 8 * g) * 32 + ln]      = (_Float16)s0[r];
      pbuf[(r + 8 * g) * 32 + 16 + ln] = (_Float16)s1[r];
    }
    __syncthreads();
    union { v16h v; _Float16 e[16]; } ap; // A-layout read: lane m=ln
#pragma unroll
    for (int t = 0; t < 8; t++) {
      ap.e[t]     = pbuf[ln * 32 + 16 * g + t];
      ap.e[8 + t] = pbuf[ln * 32 + 16 * g + 8 + t];
    }
    O0 = wmma_f16(ap.v, vb0, O0);
    O1 = wmma_f16(ap.v, vb1, O1);
  }

#pragma unroll
  for (int r = 0; r < 8; r++) {
    int srow = qt * 16 + r + 8 * g;
    if (srow < Sq) {
      float inv = 1.f / lrow[r];
      size_t base = ((size_t)(b * Sq + srow)) * 256 + hh * 32;
      out[base + ln]      = O0[r] * inv;
      out[base + 16 + ln] = O1[r] * inv;
    }
  }
}

// --------------------------- heads ------------------------------------------
__global__ void head_kernel(const float* __restrict__ x, const float* __restrict__ W,
                            const float* __restrict__ bv, float* __restrict__ out,
                            int tok, int osz) {
  int b = blockIdx.x, o = threadIdx.x;
  if (o >= osz) return;
  const float* row = x + ((size_t)(b * Sq + tok)) * 256;
  float acc = bv[o];
  for (int d = 0; d < 256; d++) acc += row[d] * W[d * osz + o];
  out[b * osz + o] = acc;
}

// ===========================================================================
extern "C" void kernel_launch(void* const* d_in, const int* in_sizes, int n_in,
                              void* d_out, int out_size, void* d_ws, size_t ws_size,
                              hipStream_t stream) {
  (void)in_sizes; (void)n_in; (void)out_size; (void)ws_size;

  // ---- input indices (jax pytree flatten: dict keys sorted) ----
  const int*   cancer = (const int*)  d_in[0];
  const int*   cand   = (const int*)  d_in[1];
  const int*   sex    = (const int*)  d_in[2];
  const float* age    = (const float*)d_in[3];
  const int*   race   = (const int*)  d_in[4];
  const float* timev  = (const float*)d_in[5];
  const float* eventv = (const float*)d_in[6];
  const int*   gene_id= (const int*)  d_in[7];
  const float* gene_emb=(const float*)d_in[8];
  const float* maf    = (const float*)d_in[9];
  const int*   fc_id  = (const int*)  d_in[10];
  const int*   fc_alt = (const int*)  d_in[11];
  const float* fc_emb = (const float*)d_in[12];
  const int*   seg_id = (const int*)  d_in[13];
  const float* seg_s  = (const float*)d_in[14];
  const float* seg_e  = (const float*)d_in[15];
  const float* seg_m  = (const float*)d_in[16];
  // params (sorted keys): W_age,W_gene,W_maf,W_seg,W_surv,b_age,b_gene,b_maf,
  //   b_seg,b_surv, blocks[4]{W1,W2,Wo,Wqkv,b1,b2,bo,bqkv,ln1_b,ln1_s,ln2_b,ln2_s},
  //   emb_cancer,emb_cancer_d,emb_clin_id,emb_focal_cna,emb_gene_id,emb_race,
  //   emb_seg_id,emb_sex, heads[6]{W,b}, ln_f_b, ln_f_s
  const float* W_age  = (const float*)d_in[17];
  const float* W_gene = (const float*)d_in[18];
  const float* W_maf  = (const float*)d_in[19];
  const float* W_seg  = (const float*)d_in[20];
  const float* W_surv = (const float*)d_in[21];
  const float* b_age  = (const float*)d_in[22];
  const float* b_gene = (const float*)d_in[23];
  const float* b_maf  = (const float*)d_in[24];
  const float* b_seg  = (const float*)d_in[25];
  const float* b_surv = (const float*)d_in[26];
  const int IB = 27;                       // blocks base, 12 leaves per block
  const float* emb_cancer   = (const float*)d_in[75];
  const float* emb_cancer_d = (const float*)d_in[76];
  const float* emb_clin_id  = (const float*)d_in[77];
  const float* emb_focal    = (const float*)d_in[78];
  const float* emb_gene_id  = (const float*)d_in[79];
  const float* emb_race     = (const float*)d_in[80];
  const float* emb_seg_id   = (const float*)d_in[81];
  const float* emb_sex      = (const float*)d_in[82];
  const int HB = 83;                       // heads base, (W,b) per head
  const float* ln_f_b = (const float*)d_in[95];
  const float* ln_f_s = (const float*)d_in[96];

  // ---- workspace layout ----
  char* ws = (char*)d_ws;
  size_t off = 0;
  auto al = [&](size_t n) { size_t o = off; off = (off + n + 255) & ~(size_t)255; return o; };
  size_t oF    = al((size_t)Bc * NIg * 320 * 4);
  size_t oItem = al((size_t)Bc * NIg * 256 * 4);
  size_t oSegE = al((size_t)Bc * NSs * 256 * 4);
  size_t oPoolG= al((size_t)Bc * NIg * 256 * 4);
  size_t oPoolS= al((size_t)Bc * NSs * 256 * 4);
  size_t oX    = al((size_t)MR * 256 * 4);
  size_t oH    = al((size_t)MR * 256 * 4);
  size_t oQKV  = al((size_t)MR * 768 * 4);
  size_t oT    = al((size_t)MR * 1024 * 4);
  size_t oAtt  = al((size_t)MR * 256 * 4);
  size_t fhsz  = (size_t)Bc * Hh * SpQ * 32 * 2;
  size_t oQh   = al(fhsz);
  size_t oKh   = al(fhsz);
  size_t oVt   = al(fhsz);
  size_t oPad  = al((size_t)Bc * Sq * 4);
  size_t oGU   = al((size_t)Bc * NIg * 4);
  size_t oSU   = al((size_t)Bc * NSs * 4);
  size_t oWt   = al((size_t)(81920 + 4 * 786432) * 2);

  float* F     = (float*)(ws + oF);
  float* item  = (float*)(ws + oItem);
  float* segE  = (float*)(ws + oSegE);
  float* poolG = (float*)(ws + oPoolG);
  float* poolS = (float*)(ws + oPoolS);
  float* X     = (float*)(ws + oX);
  float* Hbuf  = (float*)(ws + oH);
  float* QKV   = (float*)(ws + oQKV);
  float* Tbuf  = (float*)(ws + oT);
  float* Att   = (float*)(ws + oAtt);
  _Float16* Qh = (_Float16*)(ws + oQh);
  _Float16* Kh = (_Float16*)(ws + oKh);
  _Float16* Vt = (_Float16*)(ws + oVt);
  int* padArr  = (int*)(ws + oPad);
  int* uidG    = (int*)(ws + oGU);
  int* uidS    = (int*)(ws + oSU);
  _Float16* wt = (_Float16*)(ws + oWt);
  _Float16* wtGene = wt;                                   // [256,320]
  auto wtQKV = [&](int l) { return wt + 81920 + (size_t)l * 786432; };           // [768,256]
  auto wtO   = [&](int l) { return wt + 81920 + (size_t)l * 786432 + 196608; };  // [256,256]
  auto wtW1  = [&](int l) { return wt + 81920 + (size_t)l * 786432 + 262144; };  // [1024,256]
  auto wtW2  = [&](int l) { return wt + 81920 + (size_t)l * 786432 + 524288; };  // [256,1024]

  auto cvt = [&](const float* W, _Float16* Wt, int K, int N) {
    int n = K * N;
    cvtT_kernel<<<(n + 255) / 256, 256, 0, stream>>>(W, Wt, K, N);
  };

  // ---- 1. weight convert/transpose (f32 [K,N] -> f16 [N,K]) ----
  cvt(W_gene, wtGene, 320, 256);
  for (int l = 0; l < NLn; l++) {
    cvt((const float*)d_in[IB + 12 * l + 3], wtQKV(l), 256, 768);  // Wqkv
    cvt((const float*)d_in[IB + 12 * l + 2], wtO(l),   256, 256);  // Wo
    cvt((const float*)d_in[IB + 12 * l + 0], wtW1(l),  256, 1024); // W1
    cvt((const float*)d_in[IB + 12 * l + 1], wtW2(l),  1024, 256); // W2
  }

  // ---- 2. item features + projection (WMMA GEMM: 12288x256x320) ----
  features_kernel<<<dim3(NIg, Bc), 320, 0, stream>>>(
      gene_emb, maf, fc_emb, fc_alt, W_maf, b_maf, emb_focal, F);
  gemm16_kernel<<<dim3(256 / 64, (Bc * NIg) / 32), 32, 0, stream>>>(
      F, wtGene, b_gene, nullptr, item, Bc * NIg, 256, 320, 0);

  // ---- 3. seg projection + pooling ----
  segproj_kernel<<<dim3(NSs, Bc), 256, 0, stream>>>(seg_s, seg_e, seg_m, W_seg, b_seg, segE);
  pool_kernel<<<Bc, 256, 0, stream>>>(item, gene_id, fc_id, 512, 256, 1024, GPAD, poolG, uidG);
  pool_kernel<<<Bc, 256, 0, stream>>>(segE, seg_id, nullptr, 512, 0, 512, SPAD, poolS, uidS);

  // ---- 4. token assembly ----
  assemble_kernel<<<dim3(Sq, Bc), 256, 0, stream>>>(
      cancer, cand, sex, age, race, timev, eventv,
      emb_cancer, emb_cancer_d, emb_sex, emb_race,
      W_age, b_age, W_surv, b_surv, emb_clin_id, emb_gene_id, emb_seg_id,
      poolG, uidG, poolS, uidS, X, padArr);

  // ---- 5. transformer blocks ----
  for (int l = 0; l < NLn; l++) {
    const float* b1   = (const float*)d_in[IB + 12 * l + 4];
    const float* b2   = (const float*)d_in[IB + 12 * l + 5];
    const float* bo   = (const float*)d_in[IB + 12 * l + 6];
    const float* bqkv = (const float*)d_in[IB + 12 * l + 7];
    const float* ln1b = (const float*)d_in[IB + 12 * l + 8];
    const float* ln1s = (const float*)d_in[IB + 12 * l + 9];
    const float* ln2b = (const float*)d_in[IB + 12 * l + 10];
    const float* ln2s = (const float*)d_in[IB + 12 * l + 11];

    ln_kernel<<<MR, 256, 0, stream>>>(X, ln1s, ln1b, Hbuf, nullptr);
    gemm16_kernel<<<dim3(768 / 64, MR / 32), 32, 0, stream>>>(
        Hbuf, wtQKV(l), bqkv, nullptr, QKV, MR, 768, 256, 0);
    repack_kernel<<<dim3(SpQ, Bc), 32, 0, stream>>>(QKV, Qh, Kh, Vt);
    attn_kernel<<<dim3((Sq + 15) / 16, Hh, Bc), 32, 0, stream>>>(Qh, Kh, Vt, padArr, Att);
    gemm16_kernel<<<dim3(256 / 64, MR / 32), 32, 0, stream>>>(
        Att, wtO(l), bo, X, X, MR, 256, 256, 0);           // x += attn @ Wo + bo
    ln_kernel<<<MR, 256, 0, stream>>>(X, ln2s, ln2b, Hbuf, nullptr);
    gemm16_kernel<<<dim3(1024 / 64, MR / 32), 32, 0, stream>>>(
        Hbuf, wtW1(l), b1, nullptr, Tbuf, MR, 1024, 256, 1); // gelu(h@W1+b1)
    gemm16_kernel<<<dim3(256 / 64, MR / 32), 32, 0, stream>>>(
        Tbuf, wtW2(l), b2, X, X, MR, 256, 1024, 0);        // x += t @ W2 + b2
  }

  // ---- 6. final LN (+pad zero) and heads ----
  ln_kernel<<<MR, 256, 0, stream>>>(X, ln_f_s, ln_f_b, Hbuf, padArr);
  const int osz[6] = {32, 128, 2, 1, 5, 1};
  int ooff = 0;
  for (int i = 0; i < 6; i++) {
    head_kernel<<<Bc, 128, 0, stream>>>(
        Hbuf, (const float*)d_in[HB + 2 * i], (const float*)d_in[HB + 2 * i + 1],
        (float*)d_out + ooff, i, osz[i]);
    ooff += Bc * osz[i];
  }
}
